// TemporalFusionTransformer_86517821210824
// MI455X (gfx1250) — matrix operand order repaired and checked by reading, not compiled, not run on hardware
//
#include <hip/hip_runtime.h>
#include <math.h>

// Problem dims (compile-time)
#define BB 32
#define TT 512
#define VV 16
#define DD 256
#define HH 4
#define LL 2
#define BT (BB*TT)   // 16384 token rows

typedef __attribute__((ext_vector_type(16))) _Float16 v16h;
typedef __attribute__((ext_vector_type(8)))  _Float16 v8h;
typedef __attribute__((ext_vector_type(8)))  float    v8f;

__device__ __forceinline__ v8f wmma16(v16h a, v16h b, v8f c) {
  // D = A(16x32 f16) * B(32x16 f16) + C(16x16 f32)
  return __builtin_amdgcn_wmma_f32_16x16x32_f16(false, a, false, b, (short)0, c, false, false);
}
__device__ __forceinline__ float sigmoidf_(float x){ return 1.f/(1.f+__expf(-x)); }
__device__ __forceinline__ float eluf_(float x){ return x>0.f ? x : (__expf(x)-1.f); }
__device__ __forceinline__ float wsum32(float x){
  #pragma unroll
  for (int o=16;o>0;o>>=1) x += __shfl_xor(x,o,32);
  return x;
}

// Build the A fragment (lane already offset by +16 halves for hi half-wave).
__device__ __forceinline__ v16h load_afrag(const _Float16* p, int k0){
  v8h lo = *(const v8h*)(p + k0);
  v8h hi = *(const v8h*)(p + k0 + 8);
  v16h a;
  #pragma unroll
  for (int i=0;i<8;++i){ a[i]=lo[i]; a[i+8]=hi[i]; }
  return a;
}

// ---------------- elementwise prep kernels ----------------
__global__ void k_cvt(const float* __restrict__ in, _Float16* __restrict__ out, int n){
  int i = blockIdx.x*blockDim.x + threadIdx.x;
  if (i < n) out[i] = (_Float16)in[i];
}
// in: [batch][K][N] f32 -> out: [batch][N][K] f16
__global__ void k_transpose_cvt(const float* __restrict__ in, _Float16* __restrict__ out,
                                int K, int N, int total){
  int i = blockIdx.x*blockDim.x + threadIdx.x;
  if (i >= total) return;
  int v = i / (N*K);
  int r = i % (N*K);
  int n = r / K, k = r % K;
  out[i] = (_Float16)in[(size_t)v*K*N + (size_t)k*N + n];
}
// embedded[bt,v,d] = x[bt,v]*emb_w[v,d]+emb_b[v,d]  (f16 for GEMM A)
__global__ void k_embed(const float* __restrict__ x, const float* __restrict__ ew,
                        const float* __restrict__ eb, _Float16* __restrict__ out, int n){
  int i = blockIdx.x*blockDim.x + threadIdx.x;
  if (i >= n) return;
  int d = i % DD;
  int v = (i / DD) % VV;
  int bt = i / (DD*VV);
  out[i] = (_Float16)(x[(size_t)bt*VV + v]*ew[v*DD+d] + eb[v*DD+d]);
}
__global__ void k_add_cvt(const float* __restrict__ a, const float* __restrict__ b,
                          float* __restrict__ o, _Float16* __restrict__ o16, int n){
  int i = blockIdx.x*blockDim.x + threadIdx.x;
  if (i >= n) return;
  float s = a[i] + b[i];
  o[i] = s; o16[i] = (_Float16)s;
}
__global__ void k_vecadd(const float* __restrict__ a, const float* __restrict__ b,
                         float* __restrict__ o, int n){
  int i = blockIdx.x*blockDim.x + threadIdx.x;
  if (i < n) o[i] = a[i] + b[i];
}
__global__ void k_gather_last(const float* __restrict__ fg, float* __restrict__ out){
  int i = blockIdx.x*blockDim.x + threadIdx.x;
  if (i >= BB*DD) return;
  int b = i / DD, d = i % DD;
  out[i] = fg[((size_t)b*TT + (TT-1))*DD + d];
}

// ---------------- WMMA GEMM: C[M,N] = act(A[M,K] * Bt[N,K]^T + bias) ----------------
// Each wave computes a 16 x (16*NT) strip: one A fragment feeds NT independent
// WMMA accumulation chains (XDL pipe stays full; A traffic / FLOP drops NTx).
template<int NT>
__global__ void k_gemm(const _Float16* __restrict__ A, int lda,
                       const _Float16* __restrict__ Bt,
                       const float* __restrict__ bias,
                       float* __restrict__ C, int ldc,
                       _Float16* __restrict__ C16, int ldc16,
                       int M, int N, int K, int act){
  int w = (blockIdx.x*blockDim.x + threadIdx.x) >> 5;
  int lane = threadIdx.x & 31;
  int ngrp = N / (16*NT);
  int total = (M >> 4) * ngrp;
  if (w >= total) return;
  int mt = w / ngrp, ng = w % ngrp;
  int laneM = lane & 15;
  bool hi = lane >= 16;
  const _Float16* arow  = A  + (size_t)(mt*16 + laneM)*lda + (hi?16:0);
  const _Float16* brow0 = Bt + (size_t)(ng*NT*16 + laneM)*K + (hi?16:0);
  v8f acc[NT];
  #pragma unroll
  for (int j=0;j<NT;++j) acc[j] = (v8f){};
  for (int k0 = 0; k0 < K; k0 += 32) {
    v16h av = load_afrag(arow, k0);
    __builtin_prefetch(brow0 + k0 + 512, 0, 3);   // global_prefetch_b8 on weight stream
    #pragma unroll
    for (int j=0;j<NT;++j) {
      v16h bv = *(const v16h*)(brow0 + (size_t)j*16*K + k0);
      acc[j] = wmma16(av, bv, acc[j]);
    }
  }
  #pragma unroll
  for (int j=0;j<NT;++j) {
    #pragma unroll
    for (int r = 0; r < 8; ++r) {
      int row = mt*16 + r + (hi?8:0);
      int col = ng*NT*16 + j*16 + laneM;
      float vv = acc[j][r] + (bias ? bias[col] : 0.f);
      if (act == 1) vv = eluf_(vv);
      C[(size_t)row*ldc + col] = vv;
      if (C16) C16[(size_t)row*ldc16 + col] = (_Float16)vv;
    }
  }
}
static inline void gemm(hipStream_t s, const _Float16* A, int lda, const _Float16* Bt,
                        const float* bias, float* C, int ldc, _Float16* C16, int ldc16,
                        int M, int N, int K, int act){
  if (N % 64 == 0) {
    int tiles = (M/16)*(N/64);
    k_gemm<4><<<(tiles+7)/8, 256, 0, s>>>(A, lda, Bt, bias, C, ldc, C16, ldc16, M, N, K, act);
  } else if (N % 32 == 0) {
    int tiles = (M/16)*(N/32);
    k_gemm<2><<<(tiles+7)/8, 256, 0, s>>>(A, lda, Bt, bias, C, ldc, C16, ldc16, M, N, K, act);
  } else {
    int tiles = (M/16)*(N/16);
    k_gemm<1><<<(tiles+7)/8, 256, 0, s>>>(A, lda, Bt, bias, C, ldc, C16, ldc16, M, N, K, act);
  }
}

// ---------------- row kernels (wave per token row) ----------------
// variable GRN tail: GLU + skip(embedded, recomputed) + LN, then vsn += w_v * out
__global__ void k_var_glu_ln_acc(const float* __restrict__ h, const float* __restrict__ x,
                                 const float* __restrict__ ew, const float* __restrict__ eb,
                                 const float* __restrict__ g, const float* __restrict__ bn,
                                 const float* __restrict__ wsel, float* __restrict__ vsn,
                                 int v, int accum){
  int wv = threadIdx.x >> 5, lane = threadIdx.x & 31;
  int row = blockIdx.x*8 + wv;
  float xv = x[(size_t)row*VV + v];
  float vals[8];
  #pragma unroll
  for (int i = 0; i < 8; ++i) {
    int d = lane + i*32;
    float h1 = h[(size_t)row*512 + d];
    float h2 = h[(size_t)row*512 + 256 + d];
    float emb = xv*ew[v*DD+d] + eb[v*DD+d];
    vals[i] = h1*sigmoidf_(h2) + emb;
  }
  float s = 0.f;
  #pragma unroll
  for (int i=0;i<8;++i) s += vals[i];
  float mean = wsum32(s) * (1.f/256.f);
  float vs = 0.f;
  #pragma unroll
  for (int i=0;i<8;++i){ float df = vals[i]-mean; vs += df*df; }
  float inv = rsqrtf(wsum32(vs)*(1.f/256.f) + 1e-5f);
  float wgt = wsel[(size_t)row*VV + v];
  #pragma unroll
  for (int i=0;i<8;++i){
    int d = lane + i*32;
    float o = (vals[i]-mean)*inv*g[v*DD+d] + bn[v*DD+d];
    size_t oi = (size_t)row*DD + d;
    if (accum) vsn[oi] += wgt*o; else vsn[oi] = wgt*o;
  }
}
// generic GRN tail: GLU(h[BT,512]) + skip + LN -> out(+f16)
__global__ void k_glu_skip_ln(const float* __restrict__ h, const float* __restrict__ skip,
                              const float* __restrict__ g, const float* __restrict__ bn,
                              float* __restrict__ out, _Float16* __restrict__ out16){
  int wv = threadIdx.x >> 5, lane = threadIdx.x & 31;
  int row = blockIdx.x*8 + wv;
  float vals[8];
  #pragma unroll
  for (int i=0;i<8;++i){
    int d = lane + i*32;
    float h1 = h[(size_t)row*512 + d];
    float h2 = h[(size_t)row*512 + 256 + d];
    vals[i] = h1*sigmoidf_(h2) + skip[(size_t)row*DD + d];
  }
  float s=0.f;
  #pragma unroll
  for (int i=0;i<8;++i) s += vals[i];
  float mean = wsum32(s)*(1.f/256.f);
  float vs=0.f;
  #pragma unroll
  for (int i=0;i<8;++i){ float df=vals[i]-mean; vs += df*df; }
  float inv = rsqrtf(wsum32(vs)*(1.f/256.f) + 1e-5f);
  #pragma unroll
  for (int i=0;i<8;++i){
    int d = lane + i*32;
    float o = (vals[i]-mean)*inv*g[d] + bn[d];
    out[(size_t)row*DD + d] = o;
    if (out16) out16[(size_t)row*DD + d] = (_Float16)o;
  }
}
// residual + LN -> out(+f16)
__global__ void k_residual_ln(const float* __restrict__ a, const float* __restrict__ res,
                              const float* __restrict__ g, const float* __restrict__ bn,
                              float* __restrict__ out, _Float16* __restrict__ out16){
  int wv = threadIdx.x >> 5, lane = threadIdx.x & 31;
  int row = blockIdx.x*8 + wv;
  float vals[8];
  #pragma unroll
  for (int i=0;i<8;++i){
    int d = lane + i*32;
    vals[i] = a[(size_t)row*DD + d] + res[(size_t)row*DD + d];
  }
  float s=0.f;
  #pragma unroll
  for (int i=0;i<8;++i) s += vals[i];
  float mean = wsum32(s)*(1.f/256.f);
  float vs=0.f;
  #pragma unroll
  for (int i=0;i<8;++i){ float df=vals[i]-mean; vs += df*df; }
  float inv = rsqrtf(wsum32(vs)*(1.f/256.f) + 1e-5f);
  #pragma unroll
  for (int i=0;i<8;++i){
    int d = lane + i*32;
    float o = (vals[i]-mean)*inv*g[d] + bn[d];
    out[(size_t)row*DD + d] = o;
    out16[(size_t)row*DD + d] = (_Float16)o;
  }
}
// selection GRN tail: GLU(selh[BT,32]) + skip + LN(16) + softmax(16) -> weights
__global__ void k_sel_glu_softmax(const float* __restrict__ selh, const float* __restrict__ skip,
                                  const float* __restrict__ g, const float* __restrict__ bn,
                                  float* __restrict__ wout){
  int wv = threadIdx.x >> 5, lane = threadIdx.x & 31;
  int row = blockIdx.x*8 + wv;
  if (lane < 16) {
    int d = lane;
    float h1 = selh[(size_t)row*32 + d];
    float h2 = selh[(size_t)row*32 + 16 + d];
    float val = h1*sigmoidf_(h2) + skip[(size_t)row*16 + d];
    float s = val;
    #pragma unroll
    for (int o=8;o>0;o>>=1) s += __shfl_xor(s,o,16);
    float mean = s*(1.f/16.f);
    float df = val - mean;
    float vs = df*df;
    #pragma unroll
    for (int o=8;o>0;o>>=1) vs += __shfl_xor(vs,o,16);
    float ln = df*rsqrtf(vs*(1.f/16.f) + 1e-5f)*g[d] + bn[d];
    float m = ln;
    #pragma unroll
    for (int o=8;o>0;o>>=1) m = fmaxf(m, __shfl_xor(m,o,16));
    float e = __expf(ln - m);
    float es = e;
    #pragma unroll
    for (int o=8;o>0;o>>=1) es += __shfl_xor(es,o,16);
    wout[(size_t)row*16 + d] = e/es;
  }
}

// ---------------- LSTM layer scan ----------------
// 2 workgroups (one per 16-row batch half), 32 waves each; each wave owns two
// 16-column tiles of the 1024 gate columns.
// VGPR budget at 8 waves/SIMD is ~128 regs, so FULL B residency (128 regs of B
// alone) cannot fit -> keep ONE tile's 8 B fragments (64 VGPRs) register
// resident for the whole scan, and STREAM the second tile's fragments from L2
// each step. The opaque asm on the streamed pointer stops LICM from hoisting
// those loads (which is what previously caused scratch spill/reload).
__global__ void __launch_bounds__(1024)
k_lstm_scan(const float* __restrict__ xi,      // [B,T,1024] (row bt = b*T+t)
            const _Float16* __restrict__ whht, // [1024,256] f16 (N,K)
            float* __restrict__ hseq, _Float16* __restrict__ hseq16){
  __shared__ _Float16 hS[16*DD];     //  8 KB current h (f16 WMMA A)
  __shared__ float    cS[16*DD];     // 16 KB cell state
  __shared__ float    gS[16*1024];   // 64 KB gate pre-activations
  int bh = blockIdx.x;               // batch half: rows b = bh*16 .. bh*16+15
  int tid = threadIdx.x, lane = tid & 31, wv = tid >> 5;
  int laneM = lane & 15;
  bool hi = lane >= 16;
  for (int i = tid; i < 16*DD; i += 1024){ hS[i] = (_Float16)0.f; cS[i] = 0.f; }
  // Resident B: first of this wave's two column tiles (64 VGPRs, lives all scan).
  v16h bfrag0[8];
  {
    const _Float16* brow = whht + (size_t)((wv*2 + 0)*16 + laneM)*DD + (hi?16:0);
    #pragma unroll
    for (int k = 0; k < 8; ++k) bfrag0[k] = *(const v16h*)(brow + k*32);
  }
  const _Float16* brow1_base = whht + (size_t)((wv*2 + 1)*16 + laneM)*DD + (hi?16:0);
  __syncthreads();
  const _Float16* arow = &hS[laneM*DD] + (hi?16:0);
  for (int t = 0; t < TT; ++t) {
    v8f acc[2];
    #pragma unroll
    for (int j = 0; j < 2; ++j) {
      int col = (wv*2 + j)*16 + laneM;
      #pragma unroll
      for (int r = 0; r < 8; ++r) {
        int m = r + (hi?8:0);
        acc[j][r] = xi[((size_t)(bh*16 + m)*TT + t)*1024 + col];
      }
    }
    // Streamed B pointer: made opaque each iteration so loads are re-issued
    // from L2 instead of being hoisted (and spilled) by LICM.
    const _Float16* bp1 = brow1_base;
    asm volatile("" : "+v"(bp1));
    #pragma unroll
    for (int k = 0; k < 8; ++k) {
      v16h av = load_afrag(arow, k*32);
      acc[0] = wmma16(av, bfrag0[k], acc[0]);
      v16h bv = *(const v16h*)(bp1 + k*32);
      acc[1] = wmma16(av, bv, acc[1]);
    }
    #pragma unroll
    for (int j = 0; j < 2; ++j) {
      int col = (wv*2 + j)*16 + laneM;
      #pragma unroll
      for (int r = 0; r < 8; ++r) {
        int m = r + (hi?8:0);
        gS[m*1024 + col] = acc[j][r];
      }
    }
    __syncthreads();   // gates complete; h reads for step t complete
    #pragma unroll
    for (int u = 0; u < 4; ++u) {
      int idx = tid + u*1024;        // 0..4095 -> (m,d)
      int m = idx >> 8, d = idx & 255;
      float ig = sigmoidf_(gS[m*1024 + d]);
      float fg = sigmoidf_(gS[m*1024 + 256 + d]);
      float gg = tanhf   (gS[m*1024 + 512 + d]);
      float og = sigmoidf_(gS[m*1024 + 768 + d]);
      float c = fg*cS[idx] + ig*gg;
      cS[idx] = c;
      float hh = og*tanhf(c);
      hS[idx] = (_Float16)hh;
      size_t o = ((size_t)(bh*16 + m)*TT + t)*DD + d;
      hseq[o] = hh;
      hseq16[o] = (_Float16)hh;
    }
    __syncthreads();   // h/c updated before next step reads
  }
}

// ---------------- attention ----------------
// qkv f32 [BT,768] -> q16/k16 [B,H,T,64], v16t [B,H,64,T]; scale folded into q
__global__ void k_qkv_reshape(const float* __restrict__ qkv, _Float16* __restrict__ q16,
                              _Float16* __restrict__ k16, _Float16* __restrict__ v16t){
  int i = blockIdx.x*blockDim.x + threadIdx.x;
  const int n = BB*HH*TT*64;
  if (i >= n) return;
  int j = i % 64;
  int t = (i / 64) % TT;
  int h = (i / (64*TT)) % HH;
  int b = i / (64*TT*HH);
  size_t bt = (size_t)b*TT + t;
  const float* base = qkv + bt*768 + h*64 + j;
  q16[i] = (_Float16)(base[0]*0.125f);       // 1/sqrt(64)
  k16[i] = (_Float16)(base[256]);
  v16t[((size_t)(b*HH + h)*64 + j)*TT + t] = (_Float16)(base[512]);
}
// one wave per (b,h,16-query tile): S=QK^T (WMMA) -> softmax in LDS -> O=P V (WMMA)
__global__ void k_attn(const _Float16* __restrict__ q, const _Float16* __restrict__ k,
                       const _Float16* __restrict__ vt, _Float16* __restrict__ ao16){
  __shared__ float    sc[16*TT];     // 32 KB scores
  __shared__ _Float16 p16[16*TT];    // 16 KB probs
  int qt = blockIdx.x % (TT/16);
  int h  = (blockIdx.x/(TT/16)) % HH;
  int b  = blockIdx.x/((TT/16)*HH);
  int lane = threadIdx.x, laneM = lane & 15;
  bool hi = lane >= 16;
  const _Float16* qrow = q + ((size_t)(b*HH + h)*TT + qt*16 + laneM)*64 + (hi?16:0);
  const _Float16* kbase = k + (size_t)(b*HH + h)*TT*64;
  // Q fragment is invariant over all key tiles: load once.
  v16h qa0 = load_afrag(qrow, 0);
  v16h qa1 = load_afrag(qrow, 32);
  for (int nt = 0; nt < TT/16; ++nt) {
    const _Float16* krow = kbase + (size_t)(nt*16 + laneM)*64 + (hi?16:0);
    v16h bv0 = *(const v16h*)(krow);
    v16h bv1 = *(const v16h*)(krow + 32);
    v8f acc = {};
    acc = wmma16(qa0, bv0, acc);
    acc = wmma16(qa1, bv1, acc);
    #pragma unroll
    for (int r = 0; r < 8; ++r)
      sc[(r + (hi?8:0))*TT + nt*16 + laneM] = acc[r];
  }
  __syncthreads();
  if (lane < 16) {
    float mx = -3.0e38f;
    for (int j = 0; j < TT; ++j) mx = fmaxf(mx, sc[lane*TT + j]);
    float s = 0.f;
    for (int j = 0; j < TT; ++j) { float e = __expf(sc[lane*TT + j] - mx); s += e; sc[lane*TT + j] = e; }
    float invs = 1.f/s;
    for (int j = 0; j < TT; ++j) p16[lane*TT + j] = (_Float16)(sc[lane*TT + j]*invs);
  }
  __syncthreads();
  const _Float16* vbase = vt + (size_t)(b*HH + h)*64*TT;
  const _Float16* prow = &p16[laneM*TT] + (hi?16:0);
  // One P fragment (from LDS) feeds 4 independent output-tile chains.
  v8f oacc[4];
  #pragma unroll
  for (int ot = 0; ot < 4; ++ot) oacc[ot] = (v8f){};
  for (int k0 = 0; k0 < TT; k0 += 32) {
    v16h av = load_afrag(prow, k0);
    #pragma unroll
    for (int ot = 0; ot < 4; ++ot) {
      v16h bv = *(const v16h*)(vbase + (size_t)(ot*16 + laneM)*TT + (hi?16:0) + k0);
      oacc[ot] = wmma16(av, bv, oacc[ot]);
    }
  }
  #pragma unroll
  for (int ot = 0; ot < 4; ++ot) {
    #pragma unroll
    for (int r = 0; r < 8; ++r) {
      int t = qt*16 + r + (hi?8:0);
      int col = h*64 + ot*16 + laneM;
      ao16[((size_t)b*TT + t)*DD + col] = (_Float16)oacc[ot][r];
    }
  }
}

// ---------------- host orchestration ----------------
extern "C" void kernel_launch(void* const* d_in, const int* in_sizes, int n_in,
                              void* d_out, int out_size, void* d_ws, size_t ws_size,
                              hipStream_t stream){
  // inputs in setup_inputs() dict order
  const float* x         = (const float*)d_in[0];
  const float* emb_w     = (const float*)d_in[1];
  const float* emb_b     = (const float*)d_in[2];
  const float* vg_fc1_w  = (const float*)d_in[3];
  const float* vg_fc1_b  = (const float*)d_in[4];
  const float* vg_fc2_w  = (const float*)d_in[5];
  const float* vg_fc2_b  = (const float*)d_in[6];
  const float* vg_ln_g   = (const float*)d_in[7];
  const float* vg_ln_b   = (const float*)d_in[8];
  const float* sel_fc1_w = (const float*)d_in[9];
  const float* sel_fc1_b = (const float*)d_in[10];
  const float* sel_fc2_w = (const float*)d_in[11];
  const float* sel_fc2_b = (const float*)d_in[12];
  const float* sel_skip_w= (const float*)d_in[13];
  const float* sel_skip_b= (const float*)d_in[14];
  const float* sel_ln_g  = (const float*)d_in[15];
  const float* sel_ln_b  = (const float*)d_in[16];
  const float* lstm_wih  = (const float*)d_in[17];
  const float* lstm_whh  = (const float*)d_in[18];
  const float* lstm_bih  = (const float*)d_in[19];
  const float* lstm_bhh  = (const float*)d_in[20];
  const float* pl_fc1_w  = (const float*)d_in[21];
  const float* pl_fc1_b  = (const float*)d_in[22];
  const float* pl_fc2_w  = (const float*)d_in[23];
  const float* pl_fc2_b  = (const float*)d_in[24];
  const float* pl_ln_g   = (const float*)d_in[25];
  const float* pl_ln_b   = (const float*)d_in[26];
  const float* attn_in_w = (const float*)d_in[27];
  const float* attn_in_b = (const float*)d_in[28];
  const float* attn_out_w= (const float*)d_in[29];
  const float* attn_out_b= (const float*)d_in[30];
  const float* attn_ln_g = (const float*)d_in[31];
  const float* attn_ln_b = (const float*)d_in[32];
  const float* pa_fc1_w  = (const float*)d_in[33];
  const float* pa_fc1_b  = (const float*)d_in[34];
  const float* pa_fc2_w  = (const float*)d_in[35];
  const float* pa_fc2_b  = (const float*)d_in[36];
  const float* pa_ln_g   = (const float*)d_in[37];
  const float* pa_ln_b   = (const float*)d_in[38];
  const float* fg_fc1_w  = (const float*)d_in[39];
  const float* fg_fc1_b  = (const float*)d_in[40];
  const float* fg_fc2_w  = (const float*)d_in[41];
  const float* fg_fc2_b  = (const float*)d_in[42];
  const float* fg_ln_g   = (const float*)d_in[43];
  const float* fg_ln_b   = (const float*)d_in[44];

  float* out_final   = (float*)d_out;            // [B,D] = 8192
  float* out_weights = (float*)d_out + BB*DD;    // [B,T,V] = 262144

  // workspace bump allocator (256B aligned)
  size_t off = 0;
  auto alloc = [&](size_t bytes)->void*{
    void* p = (char*)d_ws + off;
    off += (bytes + 255) & ~(size_t)255;
    return p;
  };
  // f16 weights, pre-transposed to [N,K] where needed
  _Float16* vg_fc1_wt  = (_Float16*)alloc((size_t)VV*DD*DD*2);
  _Float16* vg_fc2_wt  = (_Float16*)alloc((size_t)VV*DD*2*DD*2);
  _Float16* sel_fc1_wt = (_Float16*)alloc((size_t)DD*VV*DD*2);
  _Float16* sel_fc2_wt = (_Float16*)alloc((size_t)32*DD*2);
  _Float16* sel_skip_wt= (_Float16*)alloc((size_t)16*VV*DD*2);
  _Float16* wih16      = (_Float16*)alloc((size_t)LL*1024*DD*2);
  _Float16* whh16      = (_Float16*)alloc((size_t)LL*1024*DD*2);
  _Float16* plw1t      = (_Float16*)alloc((size_t)DD*DD*2);
  _Float16* plw2t      = (_Float16*)alloc((size_t)512*DD*2);
  _Float16* paw1t      = (_Float16*)alloc((size_t)DD*DD*2);
  _Float16* paw2t      = (_Float16*)alloc((size_t)512*DD*2);
  _Float16* fgw1t      = (_Float16*)alloc((size_t)1024*DD*2);
  _Float16* fgw2t      = (_Float16*)alloc((size_t)512*1024*2);
  _Float16* ain16      = (_Float16*)alloc((size_t)768*DD*2);
  _Float16* aout16     = (_Float16*)alloc((size_t)DD*DD*2);
  float*    lstm_b     = (float*)   alloc((size_t)LL*1024*4);
  // activations
  _Float16* emb16   = (_Float16*)alloc((size_t)BT*VV*DD*2);   // also flat [BT,4096] for sel GRN
  float*    S1      = (float*)   alloc((size_t)BT*1024*4);    // generic f32 GEMM out
  _Float16* S2      = (_Float16*)alloc((size_t)BT*1024*2);    // generic f16 act
  float*    selh    = (float*)   alloc((size_t)BT*32*4);
  float*    selskip = (float*)   alloc((size_t)BT*16*4);
  float*    vsn     = (float*)   alloc((size_t)BT*DD*4);
  _Float16* vsn16   = (_Float16*)alloc((size_t)BT*DD*2);
  float*    hseq    = (float*)   alloc((size_t)BT*DD*4);
  _Float16* hseq16  = (_Float16*)alloc((size_t)BT*DD*2);
  float*    plin    = (float*)   alloc((size_t)BT*DD*4);
  _Float16* plin16  = (_Float16*)alloc((size_t)BT*DD*2);
  float*    lf      = (float*)   alloc((size_t)BT*DD*4);
  _Float16* lf16    = (_Float16*)alloc((size_t)BT*DD*2);
  _Float16* q16     = (_Float16*)alloc((size_t)BB*HH*TT*64*2);
  _Float16* k16     = (_Float16*)alloc((size_t)BB*HH*TT*64*2);
  _Float16* v16t    = (_Float16*)alloc((size_t)BB*HH*64*TT*2);
  _Float16* ao16    = (_Float16*)alloc((size_t)BT*DD*2);
  (void)ws_size; (void)in_sizes; (void)n_in; (void)out_size;

  auto g1 = [](int n){ return dim3((n + 255)/256); };

  // --- weight prep ---
  k_transpose_cvt<<<g1(VV*DD*DD),256,0,stream>>>(vg_fc1_w, vg_fc1_wt, DD, DD, VV*DD*DD);
  k_transpose_cvt<<<g1(VV*DD*2*DD),256,0,stream>>>(vg_fc2_w, vg_fc2_wt, DD, 2*DD, VV*DD*2*DD);
  k_transpose_cvt<<<g1(DD*VV*DD),256,0,stream>>>(sel_fc1_w, sel_fc1_wt, VV*DD, DD, DD*VV*DD);
  k_transpose_cvt<<<g1(32*DD),256,0,stream>>>(sel_fc2_w, sel_fc2_wt, DD, 32, 32*DD);
  k_transpose_cvt<<<g1(16*VV*DD),256,0,stream>>>(sel_skip_w, sel_skip_wt, VV*DD, VV, 16*VV*DD);
  k_cvt<<<g1(LL*1024*DD),256,0,stream>>>(lstm_wih, wih16, LL*1024*DD);   // already [N,K]
  k_cvt<<<g1(LL*1024*DD),256,0,stream>>>(lstm_whh, whh16, LL*1024*DD);   // already [N,K]
  k_transpose_cvt<<<g1(DD*DD),256,0,stream>>>(pl_fc1_w, plw1t, DD, DD, DD*DD);
  k_transpose_cvt<<<g1(512*DD),256,0,stream>>>(pl_fc2_w, plw2t, DD, 512, 512*DD);
  k_transpose_cvt<<<g1(DD*DD),256,0,stream>>>(pa_fc1_w, paw1t, DD, DD, DD*DD);
  k_transpose_cvt<<<g1(512*DD),256,0,stream>>>(pa_fc2_w, paw2t, DD, 512, 512*DD);
  k_transpose_cvt<<<g1(1024*DD),256,0,stream>>>(fg_fc1_w, fgw1t, DD, 1024, 1024*DD);
  k_transpose_cvt<<<g1(512*1024),256,0,stream>>>(fg_fc2_w, fgw2t, 1024, 512, 512*1024);
  k_cvt<<<g1(768*DD),256,0,stream>>>(attn_in_w, ain16, 768*DD);          // already [N,K]
  k_cvt<<<g1(DD*DD),256,0,stream>>>(attn_out_w, aout16, DD*DD);          // already [N,K]
  k_vecadd<<<g1(LL*1024),256,0,stream>>>(lstm_bih, lstm_bhh, lstm_b, LL*1024);

  // --- embedding (f16) ---
  k_embed<<<g1(BT*VV*DD),256,0,stream>>>(x, emb_w, emb_b, emb16, BT*VV*DD);

  // --- selection GRN -> softmax weights (written to d_out tail) ---
  gemm(stream, emb16, VV*DD, sel_fc1_wt, sel_fc1_b, S1, DD, S2, DD, BT, DD, VV*DD, 1);
  gemm(stream, S2, DD, sel_fc2_wt, sel_fc2_b, selh, 32, nullptr, 0, BT, 32, DD, 0);
  gemm(stream, emb16, VV*DD, sel_skip_wt, sel_skip_b, selskip, 16, nullptr, 0, BT, 16, VV*DD, 0);
  k_sel_glu_softmax<<<BT/8,256,0,stream>>>(selh, selskip, sel_ln_g, sel_ln_b, out_weights);

  // --- per-variable GRNs, fused with weighted sum into vsn ---
  for (int v = 0; v < VV; ++v) {
    gemm(stream, emb16 + v*DD, VV*DD, vg_fc1_wt + (size_t)v*DD*DD, vg_fc1_b + v*DD,
         S1, DD, S2, DD, BT, DD, DD, 1);
    gemm(stream, S2, DD, vg_fc2_wt + (size_t)v*DD*2*DD, vg_fc2_b + v*2*DD,
         S1, 512, nullptr, 0, BT, 512, DD, 0);
    k_var_glu_ln_acc<<<BT/8,256,0,stream>>>(S1, x, emb_w, emb_b, vg_ln_g, vg_ln_b,
                                            out_weights, vsn, v, v > 0 ? 1 : 0);
  }
  k_cvt<<<g1(BT*DD),256,0,stream>>>(vsn, vsn16, BT*DD);

  // --- 2-layer LSTM ---
  gemm(stream, vsn16, DD, wih16, lstm_b, S1, 1024, nullptr, 0, BT, 1024, DD, 0);
  k_lstm_scan<<<2,1024,0,stream>>>(S1, whh16, hseq, hseq16);
  gemm(stream, hseq16, DD, wih16 + (size_t)1024*DD, lstm_b + 1024, S1, 1024, nullptr, 0, BT, 1024, DD, 0);
  k_lstm_scan<<<2,1024,0,stream>>>(S1, whh16 + (size_t)1024*DD, hseq, hseq16);

  // --- post-LSTM GRN (input = hseq + vsn) ---
  k_add_cvt<<<g1(BT*DD),256,0,stream>>>(hseq, vsn, plin, plin16, BT*DD);
  gemm(stream, plin16, DD, plw1t, pl_fc1_b, S1, DD, S2, DD, BT, DD, DD, 1);
  gemm(stream, S2, DD, plw2t, pl_fc2_b, S1, 512, nullptr, 0, BT, 512, DD, 0);
  k_glu_skip_ln<<<BT/8,256,0,stream>>>(S1, plin, pl_ln_g, pl_ln_b, lf, lf16);

  // --- multi-head self-attention ---
  gemm(stream, lf16, DD, ain16, attn_in_b, S1, 768, nullptr, 0, BT, 768, DD, 0);
  k_qkv_reshape<<<g1(BB*HH*TT*64),256,0,stream>>>(S1, q16, k16, v16t);
  k_attn<<<BB*HH*(TT/16),32,0,stream>>>(q16, k16, v16t, ao16);
  gemm(stream, ao16, DD, aout16, attn_out_b, S1, DD, nullptr, 0, BT, DD, DD, 0);
  k_residual_ln<<<BT/8,256,0,stream>>>(S1, lf, attn_ln_g, attn_ln_b, plin, plin16); // attn_out

  // --- post-attention GRN ---
  gemm(stream, plin16, DD, paw1t, pa_fc1_b, S1, DD, S2, DD, BT, DD, DD, 1);
  gemm(stream, S2, DD, paw2t, pa_fc2_b, S1, 512, nullptr, 0, BT, 512, DD, 0);
  k_glu_skip_ln<<<BT/8,256,0,stream>>>(S1, plin, pa_ln_g, pa_ln_b, lf, lf16);       // attn_filtered

  // --- final GRN ---
  gemm(stream, lf16, DD, fgw1t, fg_fc1_b, S1, 1024, S2, 1024, BT, 1024, DD, 1);
  gemm(stream, S2, 1024, fgw2t, fg_fc2_b, S1, 512, nullptr, 0, BT, 512, 1024, 0);
  k_glu_skip_ln<<<BT/8,256,0,stream>>>(S1, lf, fg_ln_g, fg_ln_b, plin, plin16);     // out

  // --- gather out[:, -1, :] ---
  k_gather_last<<<g1(BB*DD),256,0,stream>>>(plin, out_final);
}
